// NCELoss_9139690406087
// MI455X (gfx1250) — compile-verified, matrix operand order checked by reading, not run on hardware
//
#include <hip/hip_runtime.h>
#include <hip/hip_bf16.h>

// ---------------------------------------------------------------------------
// NCE loss for MI455X (gfx1250, wave32).
// One wave per input row. 51 candidate logits computed via
// V_WMMA_F32_16X16X32_F16: N-dim = 16 candidate slots (4 tiles -> 64 slots),
// K-dim = 32 E-elements per WMMA, input row lives in M=0 of A.
// Memory-bound (~836 MB of gathered f32 weight rows, largely L2-resident).
// Inner loop is explicitly double-buffered so ~20 B128 loads stay in flight
// while cvt+WMMA runs on the other buffer (avoids s_wait_loadcnt 0 per load).
// ---------------------------------------------------------------------------

typedef __attribute__((ext_vector_type(16))) _Float16 v16h;
typedef __attribute__((ext_vector_type(8)))  float    v8f;
typedef __attribute__((ext_vector_type(2)))  __fp16   h2_t;   // cvt_pkrtz result type

#define EDIM            1024
#define NROWS           4096
#define KNOISE          50
#define NCAND           51
#define NORMT           9.0f
#define WAVES_PER_BLOCK 8
#define NBLOCKS         (NROWS / WAVES_PER_BLOCK)   // 512

// Pack 16 consecutive f32 values (as 4x float4) into a v16h using
// v_cvt_pkrtz_f16_f32.
__device__ __forceinline__ v16h pack16(float4 a0, float4 a1, float4 a2, float4 a3) {
    h2_t p0 = __builtin_amdgcn_cvt_pkrtz(a0.x, a0.y);
    h2_t p1 = __builtin_amdgcn_cvt_pkrtz(a0.z, a0.w);
    h2_t p2 = __builtin_amdgcn_cvt_pkrtz(a1.x, a1.y);
    h2_t p3 = __builtin_amdgcn_cvt_pkrtz(a1.z, a1.w);
    h2_t p4 = __builtin_amdgcn_cvt_pkrtz(a2.x, a2.y);
    h2_t p5 = __builtin_amdgcn_cvt_pkrtz(a2.z, a2.w);
    h2_t p6 = __builtin_amdgcn_cvt_pkrtz(a3.x, a3.y);
    h2_t p7 = __builtin_amdgcn_cvt_pkrtz(a3.z, a3.w);
    v16h r;
    r[0]  = (_Float16)p0[0]; r[1]  = (_Float16)p0[1];
    r[2]  = (_Float16)p1[0]; r[3]  = (_Float16)p1[1];
    r[4]  = (_Float16)p2[0]; r[5]  = (_Float16)p2[1];
    r[6]  = (_Float16)p3[0]; r[7]  = (_Float16)p3[1];
    r[8]  = (_Float16)p4[0]; r[9]  = (_Float16)p4[1];
    r[10] = (_Float16)p5[0]; r[11] = (_Float16)p5[1];
    r[12] = (_Float16)p6[0]; r[13] = (_Float16)p6[1];
    r[14] = (_Float16)p7[0]; r[15] = (_Float16)p7[1];
    return r;
}

// One software-pipeline stage: 32 E-elements for A (input) and 4 B tiles.
struct BSet {
    float4 x0, x1, x2, x3;
    float4 b[4][4];
};

__device__ __forceinline__ void load_set(BSet& s, const float* xp,
                                         const float* const* wp, int e0) {
    s.x0 = *(const float4*)(xp + e0);
    s.x1 = *(const float4*)(xp + e0 + 4);
    s.x2 = *(const float4*)(xp + e0 + 16);
    s.x3 = *(const float4*)(xp + e0 + 20);
#pragma unroll
    for (int t = 0; t < 4; ++t) {
        s.b[t][0] = *(const float4*)(wp[t] + e0);
        s.b[t][1] = *(const float4*)(wp[t] + e0 + 4);
        s.b[t][2] = *(const float4*)(wp[t] + e0 + 16);
        s.b[t][3] = *(const float4*)(wp[t] + e0 + 20);
    }
}

__device__ __forceinline__ void compute_set(const BSet& s, v8f acc[4]) {
    v16h a = pack16(s.x0, s.x1, s.x2, s.x3);
#pragma unroll
    for (int t = 0; t < 4; ++t) {
        v16h b = pack16(s.b[t][0], s.b[t][1], s.b[t][2], s.b[t][3]);
        acc[t] = __builtin_amdgcn_wmma_f32_16x16x32_f16(
            false, a, false, b, (short)0, acc[t], false, false);
    }
}

__global__ void __launch_bounds__(256)
nce_main_kernel(const float* __restrict__ input,
                const int*   __restrict__ target,
                const int*   __restrict__ noise_samples,
                const float* __restrict__ noise,
                const float* __restrict__ weight,
                const float* __restrict__ bias,
                float*       __restrict__ block_partials) {
    __shared__ int   s_idx  [WAVES_PER_BLOCK][64];
    __shared__ float s_logit[WAVES_PER_BLOCK][64];
    __shared__ float s_part [WAVES_PER_BLOCK];

    const int wv   = threadIdx.x >> 5;
    const int lane = threadIdx.x & 31;
    const int row  = blockIdx.x * WAVES_PER_BLOCK + wv;   // grid exact: < NROWS

    int*   idxs   = s_idx[wv];
    float* logits = s_logit[wv];

    // Gather candidate indices: slot 0 = target, slots 1..50 = noise samples,
    // slots 51..63 padded with 0 (never read for loss).
    for (int c = lane; c < 64; c += 32) {
        int id = 0;
        if (c == 0)            id = target[row];
        else if (c <= KNOISE)  id = noise_samples[row * KNOISE + (c - 1)];
        idxs[c] = id;
    }

    // B-matrix (32x16 f16) lane layout: lanes 0-15 carry K = 0..7 & 16..23,
    // lanes 16-31 carry K = 8..15 & 24..31 for their candidate column.
    const int cl   = lane & 15;
    const int koff = (lane >> 4) * 8;

    const float* wp[4];
#pragma unroll
    for (int t = 0; t < 4; ++t)
        wp[t] = weight + (size_t)idxs[t * 16 + cl] * EDIM + koff;

    const float* xp = input + (size_t)row * EDIM + koff;

    v8f acc[4];
#pragma unroll
    for (int t = 0; t < 4; ++t) acc[t] = (v8f){};

    // Two-stage software pipeline over E: load one 32-wide chunk while the
    // other is converted + fed to WMMA.
    BSet s0, s1;
    load_set(s0, xp, wp, 0);
#pragma unroll 1
    for (int e0 = 0; e0 < EDIM; e0 += 64) {
        load_set(s1, xp, wp, e0 + 32);
        compute_set(s0, acc);
        if (e0 + 64 < EDIM)
            load_set(s0, xp, wp, e0 + 64);
        compute_set(s1, acc);
    }

    // D layout: VGPR0 (component 0), lanes 0-15 = row M=0, col N=lane.
    if (lane < 16) {
#pragma unroll
        for (int t = 0; t < 4; ++t) {
            int c = t * 16 + lane;
            logits[c] = acc[t][0] + bias[idxs[c]];
        }
    }

    // Per-row loss: rnn term for slot 0, noise terms for slots 1..50.
    float part = 0.0f;
    for (int c = lane; c < NCAND; c += 32) {     // c in {lane, lane+32}
        float p  = __expf(logits[c] - NORMT);
        float pn = (float)KNOISE * noise[idxs[c]];
        if (c == 0) part += __logf(p  / (p + pn));
        else        part += __logf(pn / (p + pn));
    }
#pragma unroll
    for (int off = 16; off > 0; off >>= 1)
        part += __shfl_xor(part, off, 32);

    if (lane == 0) s_part[wv] = part;
    __syncthreads();
    if (threadIdx.x == 0) {
        float s = 0.0f;
        for (int w = 0; w < WAVES_PER_BLOCK; ++w) s += s_part[w];
        block_partials[blockIdx.x] = s;
    }
}

// Deterministic final reduction of the 512 block partials (single wave,
// fixed summation order on every call).
__global__ void __launch_bounds__(32)
nce_reduce_kernel(const float* __restrict__ block_partials,
                  float*       __restrict__ out) {
    float s = 0.0f;
    for (int i = threadIdx.x; i < NBLOCKS; i += 32)
        s += block_partials[i];
#pragma unroll
    for (int off = 16; off > 0; off >>= 1)
        s += __shfl_xor(s, off, 32);
    if (threadIdx.x == 0)
        out[0] = -s / (float)NROWS;
}

extern "C" void kernel_launch(void* const* d_in, const int* in_sizes, int n_in,
                              void* d_out, int out_size, void* d_ws, size_t ws_size,
                              hipStream_t stream) {
    const float* input         = (const float*)d_in[0];
    const int*   target        = (const int*)  d_in[1];
    const int*   noise_samples = (const int*)  d_in[2];
    const float* noise         = (const float*)d_in[3];
    const float* weight        = (const float*)d_in[4];
    const float* bias          = (const float*)d_in[5];
    float*       out           = (float*)d_out;
    float*       partials      = (float*)d_ws;   // NBLOCKS floats

    nce_main_kernel<<<NBLOCKS, WAVES_PER_BLOCK * 32, 0, stream>>>(
        input, target, noise_samples, noise, weight, bias, partials);
    nce_reduce_kernel<<<1, 32, 0, stream>>>(partials, out);
}
